// wave_cell_75703093559439
// MI455X (gfx1250) — matrix operand order repaired and checked by reading, not compiled, not run on hardware
//
#include <hip/hip_runtime.h>
#include <hip/hip_bf16.h>

typedef __attribute__((ext_vector_type(16))) _Float16 v16h;
typedef __attribute__((ext_vector_type(8)))  _Float16 v8h;
typedef __attribute__((ext_vector_type(8)))  float    v8f;

// ---------------- problem geometry ----------------
constexpr int B_ = 8, H_ = 256, W_ = 256;
constexpr int NPX = B_ * H_ * W_;              // 524288 pixels
// d_out layout (floats), tuple order: outputs, H, C_new, V, V_hat
constexpr int OFF_OUT = 0;
constexpr int OFF_H   = NPX;
constexpr int OFF_C   = 2 * NPX;               // C_new is [B,2,H,W]
constexpr int OFF_V   = 4 * NPX;
constexpr int OFF_VH  = 5 * NPX;

// padded NHWC f16 activation planes (halo=1 so conv loads need no bounds checks)
constexpr int PH = H_ + 2, PW = W_ + 2;        // 258 x 258
constexpr size_t ACT1_OFF   = 0;
constexpr size_t ACT1_BYTES = (size_t)B_ * PH * PW * 32 * 2;   // 34,080,768
constexpr size_t ACT2_OFF   = ACT1_OFF + ACT1_BYTES;
constexpr size_t ACT2_BYTES = (size_t)B_ * PH * PW * 64 * 2;   // 68,161,536
constexpr size_t WF2_OFF    = ACT2_OFF + ACT2_BYTES;           // conv2 B-fragments
constexpr size_t WF2_BYTES  = (size_t)9 * 1 * 4 * 512 * 2;     // 36,864
constexpr size_t WF3_OFF    = WF2_OFF + WF2_BYTES;             // conv3 B-fragments
constexpr size_t WF3_BYTES  = (size_t)9 * 2 * 2 * 512 * 2;     // 36,864

// ================= elementwise: C_new, V, H (fused Laplacian) =================
__global__ __launch_bounds__(256)
void k_elem(const float* __restrict__ inputs, const float* __restrict__ H0,
            const float* __restrict__ C0, const float* __restrict__ c2p,
            float* __restrict__ out) {
  int idx = blockIdx.x * blockDim.x + threadIdx.x;
  if (idx >= NPX) return;
  int x = idx & 255, y = (idx >> 8) & 255, b = idx >> 16;
  float in  = inputs[idx];
  float c0  = C0[b * 131072 + (y << 8) + x];       // C0[:,0]
  float v   = in - H0[idx];
  float up  = (y > 0)   ? inputs[idx - 256] : 0.f;
  float dn  = (y < 255) ? inputs[idx + 256] : 0.f;
  float lf  = (x > 0)   ? inputs[idx - 1]   : 0.f;
  float rt  = (x < 255) ? inputs[idx + 1]   : 0.f;
  float lap = up + dn + lf + rt - 4.f * in;
  float Hv  = 2.f * in - c0 + c2p[0] * lap;
  out[OFF_H + idx] = Hv;
  out[OFF_V + idx] = v;
  out[OFF_C + b * 131072 + (y << 8) + x]          = in;   // C_new[:,0]
  out[OFF_C + b * 131072 + 65536 + (y << 8) + x]  = c0;   // C_new[:,1]
}

// ============ weight prep: shuffle w2/w3 into WMMA B-fragment layout ==========
// Fragment order: [tap t][kchunk kc][nchunk nn], 512 halves each.
// Lane L<16 holds column N=nn*16+L, K-slots 0..7,16..23; lane L>=16 holds same column
// range with K-slots 8..15,24..31 (mirrors the 16-bit A layout, lane-contiguous).
__global__ __launch_bounds__(256)
void k_wprep(const float* __restrict__ w2, const float* __restrict__ w3,
             _Float16* __restrict__ wf2, _Float16* __restrict__ wf3) {
  int tid = blockIdx.x * blockDim.x + threadIdx.x;
  constexpr int N2 = 9 * 1 * 4 * 512;   // 18432 halves
  constexpr int N3 = 9 * 2 * 2 * 512;   // 18432 halves
  if (tid < N2) {
    int f = tid;
    int h = f & 15, lane = (f >> 4) & 31, rest = f >> 9;
    int nn = rest & 3, t = rest >> 2;                       // KC=1, NN=4
    int n  = nn * 16 + (lane & 15);
    int k  = ((h & 8) ? 16 : 0) + ((lane >= 16) ? 8 : 0) + (h & 7); // cin
    wf2[f] = (_Float16)w2[(n * 32 + k) * 9 + t];            // w2[cout][cin][ky][kx]
  } else if (tid < N2 + N3) {
    int f = tid - N2;
    int h = f & 15, lane = (f >> 4) & 31, rest = f >> 9;
    int nn = rest & 1; rest >>= 1;
    int kc = rest & 1; int t = rest >> 1;                   // KC=2, NN=2
    int n  = nn * 16 + (lane & 15);
    int k  = kc * 32 + ((h & 8) ? 16 : 0) + ((lane >= 16) ? 8 : 0) + (h & 7);
    wf3[f] = (_Float16)w3[(n * 64 + k) * 9 + t];
  }
}

// ================= conv1 (1->32) scalar, writes padded NHWC f16 ===============
// Channel-group loop kept rolled: 8 channels (one v8h store) live at a time.
__global__ __launch_bounds__(256)
void k_conv1(const float* __restrict__ V, const float* __restrict__ sfp,
             const float* __restrict__ w1, const float* __restrict__ b1,
             _Float16* __restrict__ act1) {
  __shared__ float ws[288];
  __shared__ float bs[32];
  for (int i = threadIdx.x; i < 288; i += blockDim.x) ws[i] = w1[i];
  if (threadIdx.x < 32) bs[threadIdx.x] = b1[threadIdx.x];
  __syncthreads();
  int idx = blockIdx.x * blockDim.x + threadIdx.x;
  if (idx >= NPX) return;
  int x = idx & 255, y = (idx >> 8) & 255, b = idx >> 16;
  float inv_sf = 1.0f / sfp[0];
  float vv[9];
#pragma unroll
  for (int t = 0; t < 9; t++) {
    int yy = y + t / 3 - 1, xx = x + t % 3 - 1;
    vv[t] = (yy >= 0 && yy < H_ && xx >= 0 && xx < W_)
                ? V[(b << 16) + (yy << 8) + xx] * inv_sf : 0.f;
  }
  _Float16* o = act1 + ((size_t)(b * PH + y + 1) * PW + (x + 1)) * 32;
#pragma unroll 1
  for (int g = 0; g < 4; g++) {
    v8h pk;
#pragma unroll
    for (int j = 0; j < 8; j++) {
      const int c = g * 8 + j;
      float s = bs[c];
#pragma unroll
      for (int t = 0; t < 9; t++) s = fmaf(ws[c * 9 + t], vv[t], s);
      pk[j] = (_Float16)fmaxf(s, 0.f);
    }
    *(v8h*)(o + g * 8) = pk;
  }
}

// ========= implicit-GEMM conv (3x3 SAME) via v_wmma_f32_16x16x32_f16 =========
// One wave: M = 16 consecutive x-pixels, N = COUT (NN 16-wide chunks),
// K = CIN*9 as 9 taps x KC chunks of 32. 8 waves/block share LDS-staged weights.
// Outer dy loop kept rolled (unroll 1) to bound live A-fragments to one tap row
// -> lower VGPR pressure -> higher occupancy to hide L2 latency under WMMAs.
template <int CIN, int COUT>
__global__ __launch_bounds__(256)
void k_conv_wmma(const _Float16* __restrict__ actIn,
                 const _Float16* __restrict__ wfrag,
                 const float* __restrict__ bias,
                 _Float16* __restrict__ actOut) {
  constexpr int KC = CIN / 32;
  constexpr int NN = COUT / 16;
  constexpr int WBYTES = 9 * KC * NN * 1024;
  __shared__ __attribute__((aligned(32))) unsigned char smw[WBYTES];

  { // cooperative global -> LDS weight stage
    const uint4* src = (const uint4*)wfrag;
    uint4* dst = (uint4*)smw;
    for (int i = threadIdx.x; i < WBYTES / 16; i += blockDim.x) dst[i] = src[i];
  }
  __syncthreads();

  const int wave = threadIdx.x >> 5;
  const int lane = threadIdx.x & 31;
  const int p0 = (blockIdx.x * 8 + wave) * 16;   // 16-pixel tile base
  const int x0 = p0 & 255;
  const int y  = (p0 >> 8) & 255;
  const int b  = p0 >> 16;
  const int mrow = lane & 15;                    // A-matrix row = pixel
  const int hi8 = (lane >> 4) * 8;               // K sub-slot select per A layout

  const v8f vzero = {0.f, 0.f, 0.f, 0.f, 0.f, 0.f, 0.f, 0.f};
  v8f acc[NN];
#pragma unroll
  for (int nn = 0; nn < NN; nn++) acc[nn] = vzero;

  const _Float16* wsm = (const _Float16*)smw;
  // per-lane base for (dy=-1, dx=-1) tap; advance by one padded row per dy
  const _Float16* rowp =
      actIn + ((size_t)(b * PH + y) * PW + (x0 + mrow)) * CIN;

#pragma unroll 1
  for (int dyi = 0; dyi < 3; dyi++) {
#pragma unroll
    for (int dxi = 0; dxi < 3; dxi++) {
      const int t = dyi * 3 + dxi;
      const _Float16* p = rowp + dxi * CIN;
#pragma unroll
      for (int kc = 0; kc < KC; kc++) {
        const int c0 = kc * 32 + hi8;
        v8h lo = *(const v8h*)(p + c0);          // K slots 0..7 (or 8..15)
        v8h hh = *(const v8h*)(p + c0 + 16);     // K slots 16..23 (24..31)
        v16h a;
#pragma unroll
        for (int i = 0; i < 8; i++) { a[i] = lo[i]; a[8 + i] = hh[i]; }
#pragma unroll
        for (int nn = 0; nn < NN; nn++) {
          const v16h bf =
              *(const v16h*)(wsm + ((t * KC + kc) * NN + nn) * 512 + lane * 16);
          acc[nn] = __builtin_amdgcn_wmma_f32_16x16x32_f16(
              false, a, false, bf, (short)0, acc[nn], false, false);
        }
      }
    }
    rowp += (size_t)PW * CIN;                    // next dy row
  }

  // D layout: VGPR r -> M=r (lanes 0-15), M=8+r (lanes 16-31); N = lane%16
  const int ncol  = lane & 15;
  const int mbase = (lane >> 4) * 8;
  _Float16* orow =
      actOut + ((size_t)(b * PH + y + 1) * PW + (x0 + mbase + 1)) * COUT;
#pragma unroll
  for (int nn = 0; nn < NN; nn++) {
    const int n = nn * 16 + ncol;
    const float bv = bias[n];
#pragma unroll
    for (int r = 0; r < 8; r++) {
      float v = fmaxf(acc[nn][r] + bv, 0.f);     // bias + ReLU fused
      orow[(size_t)r * COUT + n] = (_Float16)v;
    }
  }
}

// ====== conv4 (32->1) scalar, fused V_hat = (.)*sf and outputs = H + V_hat ====
// dy loop kept rolled: only one tap-row (12 x 16B loads) live -> low VGPRs.
__global__ __launch_bounds__(256)
void k_conv4(const _Float16* __restrict__ act3, const float* __restrict__ w4,
             const float* __restrict__ b4, const float* __restrict__ sfp,
             float* out) {
  __shared__ float ws[288];
  for (int i = threadIdx.x; i < 288; i += blockDim.x) ws[i] = w4[i];
  __syncthreads();
  int idx = blockIdx.x * blockDim.x + threadIdx.x;
  if (idx >= NPX) return;
  int x = idx & 255, y = (idx >> 8) & 255, b = idx >> 16;
  float s = b4[0];
  const _Float16* base = act3 + ((size_t)(b * PH + y) * PW + x) * 32;
#pragma unroll 1
  for (int dy = 0; dy < 3; dy++) {
#pragma unroll
    for (int dx = 0; dx < 3; dx++) {
      const int t = dy * 3 + dx;
      const v8h* p8 = (const v8h*)(base + dx * 32);
#pragma unroll
      for (int g = 0; g < 4; g++) {
        v8h vg = p8[g];
#pragma unroll
        for (int j = 0; j < 8; j++)
          s = fmaf((float)vg[j], ws[(g * 8 + j) * 9 + t], s);
      }
    }
    base += (size_t)PW * 32;
  }
  float vh = s * sfp[0];
  out[OFF_VH + idx]  = vh;
  out[OFF_OUT + idx] = out[OFF_H + idx] + vh;
}

// ================================ launcher ====================================
extern "C" void kernel_launch(void* const* d_in, const int* in_sizes, int n_in,
                              void* d_out, int out_size, void* d_ws, size_t ws_size,
                              hipStream_t stream) {
  const float* inputs = (const float*)d_in[0];
  const float* H0     = (const float*)d_in[1];
  const float* C0     = (const float*)d_in[2];
  const float* c2     = (const float*)d_in[3];
  const float* sf     = (const float*)d_in[4];
  const float* w1     = (const float*)d_in[5];
  const float* b1     = (const float*)d_in[6];
  const float* w2     = (const float*)d_in[7];
  const float* b2     = (const float*)d_in[8];
  const float* w3     = (const float*)d_in[9];
  const float* b3     = (const float*)d_in[10];
  const float* w4     = (const float*)d_in[11];
  const float* b4     = (const float*)d_in[12];
  float* out = (float*)d_out;

  unsigned char* ws = (unsigned char*)d_ws;
  _Float16* act1 = (_Float16*)(ws + ACT1_OFF);   // also reused as act3
  _Float16* act2 = (_Float16*)(ws + ACT2_OFF);
  _Float16* wf2  = (_Float16*)(ws + WF2_OFF);
  _Float16* wf3  = (_Float16*)(ws + WF3_OFF);

  // zero padded activation planes (keeps 1-pixel halos at zero; capture-legal)
  hipMemsetAsync(ws, 0, ACT1_BYTES + ACT2_BYTES, stream);

  k_elem<<<NPX / 256, 256, 0, stream>>>(inputs, H0, C0, c2, out);
  k_wprep<<<(9 * 1 * 4 * 512 + 9 * 2 * 2 * 512) / 256, 256, 0, stream>>>(w2, w3, wf2, wf3);
  k_conv1<<<NPX / 256, 256, 0, stream>>>(out + OFF_V, sf, w1, b1, act1);
  k_conv_wmma<32, 64><<<NPX / 128, 256, 0, stream>>>(act1, wf2, b2, act2);
  k_conv_wmma<64, 32><<<NPX / 128, 256, 0, stream>>>(act2, wf3, b3, act1); // act3 in act1 buf
  k_conv4<<<NPX / 256, 256, 0, stream>>>(act1, w4, b4, sf, out);
}
